// GatedDeltaNet_2937757630855
// MI455X (gfx1250) — matrix operand order repaired and checked by reading, not compiled, not run on hardware
//
#include <hip/hip_runtime.h>
#include <hip/hip_bf16.h>

// ---------------- problem constants (Qwen3-Next GatedDeltaNet) ----------------
#define HDIM     2048
#define HV       32
#define HK       16
#define DK       128
#define DV       128
#define KWIN     4
#define KEY_DIM  (HK * DK)          // 2048
#define VAL_DIM  (HV * DV)          // 4096
#define CONV_DIM (2 * KEY_DIM + VAL_DIM) // 8192
#define BATCH    2
#define S_LEN    1024
#define NTOK     (BATCH * S_LEN)    // 2048

typedef __attribute__((ext_vector_type(16))) __bf16 v16bf;
typedef __attribute__((ext_vector_type(8)))  float  v8f;
typedef unsigned short ushort8 __attribute__((ext_vector_type(8)));
typedef float float4v __attribute__((ext_vector_type(4)));

__device__ __forceinline__ unsigned short f32_to_bf16(float f) {
    unsigned u = __builtin_bit_cast(unsigned, f);
    u += 0x7FFFu + ((u >> 16) & 1u);           // round-to-nearest-even
    return (unsigned short)(u >> 16);
}

// --------------------- operand pre-swizzle into WMMA fragment order ----------
// Fragment = 16x32 (A) or 32x16 (B) bf16 tile laid out exactly as the
// v_wmma_f32_16x16x32_bf16 source VGPRs expect (ISA 7.12.2), stored as
// [fragIdx][lane][16 bf16] = 32 contiguous bytes per lane.
// A (row-major [M,K]): lane L row = m0+(L&15); elem j -> k = k0+16*(j>>3)+(L>>4)*8+(j&7)
__global__ void swizzleA_kernel(const float* __restrict__ A, unsigned short* __restrict__ out,
                                int M, int K)
{
    int t = blockIdx.x * blockDim.x + threadIdx.x;      // one per (mtile, kt, lane)
    int total = (M >> 4) * (K >> 5) * 32;
    if (t >= total) return;
    int lane = t & 31;
    int f = t >> 5;                                     // mtile*ktiles + kt
    int ktiles = K >> 5;
    int mtile = f / ktiles, kt = f - mtile * ktiles;
    int row = mtile * 16 + (lane & 15);
    int kbase = kt * 32 + (lane >> 4) * 8;
    const float* src = A + (size_t)row * K + kbase;
    float4v a0 = *(const float4v*)(src);
    float4v a1 = *(const float4v*)(src + 4);
    float4v a2 = *(const float4v*)(src + 16);
    float4v a3 = *(const float4v*)(src + 20);
    ushort8 o0, o1;
#pragma unroll
    for (int i = 0; i < 4; ++i) {
        o0[i] = f32_to_bf16(a0[i]); o0[4 + i] = f32_to_bf16(a1[i]);
        o1[i] = f32_to_bf16(a2[i]); o1[4 + i] = f32_to_bf16(a3[i]);
    }
    unsigned short* dst = out + ((size_t)f * 32 + lane) * 16;
    *(ushort8*)dst = o0;
    *(ushort8*)(dst + 8) = o1;
}

// B (row-major [K,N]): lane L col = n0+(L&15); elem j -> k = k0+16*(j>>3)+(L>>4)*8+(j&7)
__global__ void swizzleB_kernel(const float* __restrict__ B, unsigned short* __restrict__ out,
                                int K, int N)
{
    int t = blockIdx.x * blockDim.x + threadIdx.x;      // one per (ntile, kt, lane)
    int total = (N >> 4) * (K >> 5) * 32;
    if (t >= total) return;
    int lane = t & 31;
    int f = t >> 5;                                     // ntile*ktiles + kt
    int ktiles = K >> 5;
    int ntile = f / ktiles, kt = f - ntile * ktiles;
    int col = ntile * 16 + (lane & 15);
    int kbase = kt * 32 + (lane >> 4) * 8;
    const float* src = B + (size_t)kbase * N + col;
    ushort8 o0, o1;
#pragma unroll
    for (int i = 0; i < 8; ++i) {
        o0[i] = f32_to_bf16(src[(size_t)i * N]);
        o1[i] = f32_to_bf16(src[(size_t)(i + 16) * N]);
    }
    unsigned short* dst = out + ((size_t)f * 32 + lane) * 16;
    *(ushort8*)dst = o0;
    *(ushort8*)(dst + 8) = o1;
}

// ------------------ WMMA GEMM on pre-swizzled bf16 operands -------------------
// C[M,N] f32 = A @ B. No LDS, no barriers: per K-step each wave loads
// 2 A-frags + 4 B-frags (12x global_load_b128) and issues 8x v_wmma.
// Block = 8 waves (2M x 4N); block tile 64x256; wave tile 32x64.
// Requires M%64==0, K%32==0, N%16==0; out-of-range N tiles are masked.
__global__ __launch_bounds__(256) void gemm_swz_wmma_kernel(
    const unsigned short* __restrict__ Asw, const unsigned short* __restrict__ Bsw,
    float* __restrict__ C, int M, int N, int K)
{
    const int lane = threadIdx.x & 31;
    const int wid  = threadIdx.x >> 5;
    const int wm = wid & 1, wn = wid >> 1;            // 2 x 4 wave grid
    const int ktiles = K >> 5;
    const int ntiles = N >> 4;
    const int mt0 = blockIdx.y * 4 + wm * 2;          // 2 M-tiles per wave
    const int nt0 = blockIdx.x * 16 + wn * 4;         // 4 N-tiles per wave
    if (nt0 >= ntiles) return;                        // safe: no barriers

    const v16bf* ap[2];
    const v16bf* bp[4];
    bool nv[4];
#pragma unroll
    for (int i = 0; i < 2; ++i)
        ap[i] = (const v16bf*)Asw + (size_t)(mt0 + i) * ktiles * 32 + lane;
#pragma unroll
    for (int j = 0; j < 4; ++j) {
        int nt = nt0 + j;
        nv[j] = nt < ntiles;
        bp[j] = (const v16bf*)Bsw + (size_t)(nv[j] ? nt : (ntiles - 1)) * ktiles * 32 + lane;
    }

    v8f acc[2][4] = {};
#pragma unroll 2
    for (int kt = 0; kt < ktiles; ++kt) {
        v16bf af[2], bf[4];
#pragma unroll
        for (int i = 0; i < 2; ++i) af[i] = ap[i][(size_t)kt * 32];
#pragma unroll
        for (int j = 0; j < 4; ++j) bf[j] = bp[j][(size_t)kt * 32];
        if (kt + 4 < ktiles) {
            __builtin_prefetch(ap[0] + (size_t)(kt + 4) * 32, 0, 3);   // near-cache prefetch
            __builtin_prefetch(bp[0] + (size_t)(kt + 4) * 32, 0, 3);
        }
#pragma unroll
        for (int i = 0; i < 2; ++i)
#pragma unroll
            for (int j = 0; j < 4; ++j)
                acc[i][j] = __builtin_amdgcn_wmma_f32_16x16x32_bf16(
                    false, af[i], false, bf[j], (short)0, acc[i][j], false, false);
    }

    // C/D layout: VGPR r -> M = r + 8*(lane>>4); N = lane&15
#pragma unroll
    for (int i = 0; i < 2; ++i) {
        int rbase = (mt0 + i) * 16 + (lane >> 4) * 8;
#pragma unroll
        for (int j = 0; j < 4; ++j) {
            if (!nv[j]) continue;
            int col = (nt0 + j) * 16 + (lane & 15);
#pragma unroll
            for (int r = 0; r < 8; ++r)
                C[(size_t)(rbase + r) * N + col] = acc[i][j][r];
        }
    }
}

// ------------------- causal depthwise conv (KW=4) + SiLU ----------------------
__global__ void conv_silu_kernel(const float* __restrict__ pre, const float* __restrict__ w,
                                 const float* __restrict__ bias, float* __restrict__ out, int total)
{
    int idx = blockIdx.x * blockDim.x + threadIdx.x;
    if (idx >= total) return;
    int c = idx % CONV_DIM;
    int s = (idx / CONV_DIM) % S_LEN;
    int b = idx / (CONV_DIM * S_LEN);
    float acc = bias[c];
#pragma unroll
    for (int j = 0; j < KWIN; ++j) {
        int ss = s + j - (KWIN - 1);
        if (ss >= 0)
            acc += pre[(size_t)(b * S_LEN + ss) * CONV_DIM + c] * w[c * KWIN + j];
    }
    out[idx] = acc / (1.0f + __expf(-acc));      // SiLU
}

// ------------- per-head l2norm of q,k (+ q scale DK^-0.5) ---------------------
__global__ __launch_bounds__(256) void qknorm_kernel(const float* __restrict__ mixed,
                                                     float* __restrict__ qn, float* __restrict__ kn)
{
    int task = blockIdx.x * 8 + (threadIdx.x >> 5);   // token*HK*2 + head*2 + which
    int lane = threadIdx.x & 31;
    int which = task & 1; int t = task >> 1;
    int h = t % HK; t /= HK;                          // t = b*S + s
    const float* src = mixed + (size_t)t * CONV_DIM + which * KEY_DIM + h * DK + lane * 4;
    float e0 = src[0], e1 = src[1], e2 = src[2], e3 = src[3];
    float ss = e0 * e0 + e1 * e1 + e2 * e2 + e3 * e3;
#pragma unroll
    for (int off = 16; off >= 1; off >>= 1) ss += __shfl_xor(ss, off, 32);
    float scale = rsqrtf(ss + 1e-6f);
    if (which == 0) scale *= 0.08838834764831845f;    // DK^-0.5
    float* dst = (which == 0 ? qn : kn) + ((size_t)t * HK + h) * DK + lane * 4;
    dst[0] = e0 * scale; dst[1] = e1 * scale; dst[2] = e2 * scale; dst[3] = e3 * scale;
}

// ---------------------- beta = sigmoid, g = -exp(A)*softplus ------------------
__global__ void betag_kernel(const float* __restrict__ bpre, const float* __restrict__ apre,
                             const float* __restrict__ A_log, const float* __restrict__ dt_bias,
                             float* __restrict__ beta, float* __restrict__ g, int n)
{
    int i = blockIdx.x * blockDim.x + threadIdx.x;
    if (i >= n) return;
    int h = i % HV;
    beta[i] = 1.0f / (1.0f + __expf(-bpre[i]));
    float x = apre[i] + dt_bias[h];
    float sp = (x > 20.0f) ? x : log1pf(__expf(x));
    g[i] = -__expf(A_log[h]) * sp;
}

// ---------------- gated delta-rule recurrence (serial over S) -----------------
// One block per (b, head); thread v owns state column state[0..DK-1][v] in VGPRs.
__global__ __launch_bounds__(128) void deltarule_kernel(
    const float* __restrict__ qn, const float* __restrict__ kn,
    const float* __restrict__ mixed, const float* __restrict__ g,
    const float* __restrict__ beta, float* __restrict__ core)
{
    const int bh = blockIdx.x;                 // 0..B*HV-1
    const int b = bh / HV, h = bh % HV;
    const int hk = h >> 1;                     // rep = HV/HK = 2
    const int v = threadIdx.x;                 // 0..127
    __shared__ float qs[DK], ks[DK];

    float st[DK];
#pragma unroll
    for (int i = 0; i < DK; ++i) st[i] = 0.0f;

    for (int s = 0; s < S_LEN; ++s) {
        size_t tb = (size_t)(b * S_LEN + s);
        qs[v] = qn[(tb * HK + hk) * DK + v];
        ks[v] = kn[(tb * HK + hk) * DK + v];
        float gg = g[tb * HV + h];
        float bb = beta[tb * HV + h];
        float vt = mixed[tb * CONV_DIM + 2 * KEY_DIM + h * DV + v];
        __syncthreads();

        float decay = __expf(gg);
        float mem = 0.0f;
#pragma unroll
        for (int kk = 0; kk < DK; ++kk) { st[kk] *= decay; mem = fmaf(ks[kk], st[kk], mem); }
        float delta = (vt - mem) * bb;
        float out = 0.0f;
#pragma unroll
        for (int kk = 0; kk < DK; ++kk) { st[kk] = fmaf(ks[kk], delta, st[kk]); out = fmaf(qs[kk], st[kk], out); }
        core[(tb * HV + h) * DV + v] = out;
        __syncthreads();                        // qs/ks reused next step
    }
}

// -------------------- gated RMSNorm: core * silu(z), rms over DV --------------
__global__ __launch_bounds__(256) void rmsnorm_kernel(const float* __restrict__ core,
                                                      const float* __restrict__ z,
                                                      const float* __restrict__ nw,
                                                      float* __restrict__ out)
{
    int task = blockIdx.x * 8 + (threadIdx.x >> 5);   // (b*S+s)*HV + h
    int lane = threadIdx.x & 31;
    size_t base = (size_t)task * DV + lane * 4;
    float gv[4]; float ss = 0.0f;
#pragma unroll
    for (int i = 0; i < 4; ++i) {
        float zz = z[base + i];
        float gate = zz / (1.0f + __expf(-zz));       // silu(z)
        gv[i] = core[base + i] * gate;
        ss += gv[i] * gv[i];
    }
#pragma unroll
    for (int off = 16; off >= 1; off >>= 1) ss += __shfl_xor(ss, off, 32);
    float scale = rsqrtf(ss * (1.0f / DV) + 1e-6f);
#pragma unroll
    for (int i = 0; i < 4; ++i) out[base + i] = gv[i] * scale * nw[lane * 4 + i];
}

// ------------------------------ workspace layout ------------------------------
static const size_t OFF_XSW    = 0;                                               // 8 MB
static const size_t OFF_WQKVS  = OFF_XSW   + (size_t)NTOK * HDIM * 2;             // 33.5 MB
static const size_t OFF_WZS    = OFF_WQKVS + (size_t)HDIM * CONV_DIM * 2;         // 16.8 MB
static const size_t OFF_WBS    = OFF_WZS   + (size_t)HDIM * VAL_DIM * 2;          // 128 KB
static const size_t OFF_WAS    = OFF_WBS   + (size_t)HDIM * HV * 2;               // 128 KB
static const size_t OFF_WOUTS  = OFF_WAS   + (size_t)HDIM * HV * 2;               // 16.8 MB
static const size_t OFF_QKV    = OFF_WOUTS + (size_t)VAL_DIM * HDIM * 2;          // 64 MB
static const size_t OFF_MIXED  = OFF_QKV   + (size_t)NTOK * CONV_DIM * 4;         // 64 MB
static const size_t OFF_Z      = OFF_MIXED + (size_t)NTOK * CONV_DIM * 4;         // 32 MB
static const size_t OFF_BPRE   = OFF_Z     + (size_t)NTOK * VAL_DIM * 4;
static const size_t OFF_APRE   = OFF_BPRE  + (size_t)NTOK * HV * 4;
static const size_t OFF_BETA   = OFF_APRE  + (size_t)NTOK * HV * 4;
static const size_t OFF_G      = OFF_BETA  + (size_t)NTOK * HV * 4;
static const size_t OFF_QN     = OFF_G     + (size_t)NTOK * HV * 4;               // 16.8 MB
static const size_t OFF_KN     = OFF_QN    + (size_t)NTOK * HK * DK * 4;          // 16.8 MB
static const size_t OFF_CORE   = OFF_KN    + (size_t)NTOK * HK * DK * 4;          // 32 MB
// normed f32 (32MB) + normed swizzled bf16 (16.8MB) both alias the 64MB qkv_pre
// region, which is dead after the conv kernel consumes it.

extern "C" void kernel_launch(void* const* d_in, const int* in_sizes, int n_in,
                              void* d_out, int out_size, void* d_ws, size_t ws_size,
                              hipStream_t stream) {
    const float* x        = (const float*)d_in[0];   // [B,S,H]
    const float* Wqkv     = (const float*)d_in[1];   // [H, CONV_DIM]
    const float* Wz       = (const float*)d_in[2];   // [H, VAL_DIM]
    const float* Wb       = (const float*)d_in[3];   // [H, HV]
    const float* Wa       = (const float*)d_in[4];   // [H, HV]
    const float* conv_w   = (const float*)d_in[5];   // [CONV_DIM, KW]
    const float* conv_b   = (const float*)d_in[6];   // [CONV_DIM]
    const float* dt_bias  = (const float*)d_in[7];   // [HV]
    const float* A_log    = (const float*)d_in[8];   // [HV]
    const float* norm_w   = (const float*)d_in[9];   // [DV]
    const float* Wout     = (const float*)d_in[10];  // [VAL_DIM, H]
    (void)in_sizes; (void)n_in; (void)out_size; (void)ws_size;

    char* ws = (char*)d_ws;
    unsigned short* xsw    = (unsigned short*)(ws + OFF_XSW);
    unsigned short* wqkvs  = (unsigned short*)(ws + OFF_WQKVS);
    unsigned short* wzs    = (unsigned short*)(ws + OFF_WZS);
    unsigned short* wbs    = (unsigned short*)(ws + OFF_WBS);
    unsigned short* was    = (unsigned short*)(ws + OFF_WAS);
    unsigned short* wouts  = (unsigned short*)(ws + OFF_WOUTS);
    float* qkv_pre = (float*)(ws + OFF_QKV);
    float* mixed   = (float*)(ws + OFF_MIXED);
    float* zbuf    = (float*)(ws + OFF_Z);
    float* bpre    = (float*)(ws + OFF_BPRE);
    float* apre    = (float*)(ws + OFF_APRE);
    float* beta    = (float*)(ws + OFF_BETA);
    float* g       = (float*)(ws + OFF_G);
    float* qn      = (float*)(ws + OFF_QN);
    float* kn      = (float*)(ws + OFF_KN);
    float* core    = (float*)(ws + OFF_CORE);
    float* normed  = (float*)(ws + OFF_QKV);                          // alias (32MB)
    unsigned short* normsw = (unsigned short*)(ws + OFF_QKV + (size_t)NTOK * VAL_DIM * 4);
    float* out     = (float*)d_out;

    // 0) one-time operand swizzles into WMMA fragment order (f32 -> bf16)
    auto gridFor = [](size_t total) { return (unsigned)((total + 255) / 256); };
    swizzleA_kernel<<<gridFor((size_t)NTOK * HDIM / 16), 256, 0, stream>>>(x, xsw, NTOK, HDIM);
    swizzleB_kernel<<<gridFor((size_t)HDIM * CONV_DIM / 16), 256, 0, stream>>>(Wqkv, wqkvs, HDIM, CONV_DIM);
    swizzleB_kernel<<<gridFor((size_t)HDIM * VAL_DIM / 16), 256, 0, stream>>>(Wz, wzs, HDIM, VAL_DIM);
    swizzleB_kernel<<<gridFor((size_t)HDIM * HV / 16), 256, 0, stream>>>(Wb, wbs, HDIM, HV);
    swizzleB_kernel<<<gridFor((size_t)HDIM * HV / 16), 256, 0, stream>>>(Wa, was, HDIM, HV);
    swizzleB_kernel<<<gridFor((size_t)VAL_DIM * HDIM / 16), 256, 0, stream>>>(Wout, wouts, VAL_DIM, HDIM);

    // 1) input projections (pure WMMA GEMMs, no LDS/barriers); grid.x = ceil(N/256)
    gemm_swz_wmma_kernel<<<dim3((CONV_DIM + 255) / 256, NTOK / 64), 256, 0, stream>>>(xsw, wqkvs, qkv_pre, NTOK, CONV_DIM, HDIM);
    gemm_swz_wmma_kernel<<<dim3((VAL_DIM  + 255) / 256, NTOK / 64), 256, 0, stream>>>(xsw, wzs,   zbuf,    NTOK, VAL_DIM,  HDIM);
    gemm_swz_wmma_kernel<<<dim3(1,                      NTOK / 64), 256, 0, stream>>>(xsw, wbs,   bpre,    NTOK, HV,       HDIM);
    gemm_swz_wmma_kernel<<<dim3(1,                      NTOK / 64), 256, 0, stream>>>(xsw, was,   apre,    NTOK, HV,       HDIM);

    // 2) causal depthwise conv + SiLU
    int convTotal = NTOK * CONV_DIM;
    conv_silu_kernel<<<(convTotal + 255) / 256, 256, 0, stream>>>(qkv_pre, conv_w, conv_b, mixed, convTotal);

    // 3) l2norm q,k ; beta/g activations
    qknorm_kernel<<<(NTOK * HK * 2) / 8, 256, 0, stream>>>(mixed, qn, kn);
    int nbg = NTOK * HV;
    betag_kernel<<<(nbg + 255) / 256, 256, 0, stream>>>(bpre, apre, A_log, dt_bias, beta, g, nbg);

    // 4) gated delta-rule recurrence (serial over S, parallel over B*HV and DV)
    deltarule_kernel<<<BATCH * HV, 128, 0, stream>>>(qn, kn, mixed, g, beta, core);

    // 5) gated RMSNorm -> normed (f32), then swizzle for the output GEMM
    rmsnorm_kernel<<<(NTOK * HV) / 8, 256, 0, stream>>>(core, zbuf, norm_w, normed);
    swizzleA_kernel<<<gridFor((size_t)NTOK * VAL_DIM / 16), 256, 0, stream>>>(normed, normsw, NTOK, VAL_DIM);

    // 6) output projection (WMMA GEMM)
    gemm_swz_wmma_kernel<<<dim3((HDIM + 255) / 256, NTOK / 64), 256, 0, stream>>>(normsw, wouts, out, NTOK, HDIM, VAL_DIM);
}
